// CrossModalAttention_50749333570241
// MI455X (gfx1250) — matrix operand order repaired and checked by reading, not compiled, not run on hardware
//
#include <hip/hip_runtime.h>

// ---------------------------------------------------------------------------
// CDNA5 (gfx1250) transformer block: cross-attention + FFN, f16 WMMA GEMMs.
// Wave32. Matrix ops: v_wmma_f32_16x16x32_f16. A-tile staging via the
// Tensor Data Mover (tensor_load_to_lds, TENSORcnt) with LDS padding,
// double-buffered against WMMA compute.
// ---------------------------------------------------------------------------

typedef __attribute__((ext_vector_type(16))) _Float16     v16h;
typedef __attribute__((ext_vector_type(8)))  float        v8f;
typedef __attribute__((ext_vector_type(4)))  unsigned int u32x4;
typedef __attribute__((ext_vector_type(8)))  int          i32x8;
typedef __attribute__((ext_vector_type(4)))  int          i32x4;

#if __has_builtin(__builtin_amdgcn_tensor_load_to_lds) && \
    __has_builtin(__builtin_amdgcn_s_wait_tensorcnt)
#define USE_TDM 1
#else
#define USE_TDM 0
#endif

union HFrag { v16h v; u32x4 q[2]; };

// A-operand fragment (16x32 f16 tile), per ISA 7.12.2:
// lane row = lane&15, half = lane>>4; elems 0..7 at K = half*8+0..7,
// elems 8..15 at K = 16+half*8+0..7. rowptr points at K=0 of this lane's row.
__device__ __forceinline__ v16h frag_A(const _Float16* rowptr, int k0, int half) {
  HFrag f;
  f.q[0] = *(const u32x4*)(rowptr + k0 + half * 8);
  f.q[1] = *(const u32x4*)(rowptr + k0 + 16 + half * 8);
  return f.v;
}

// B-operand fragment (32x16 f16 tile): lane col = lane&15, half = lane>>4;
// elems 0..7 at K = half*8+0..7, elems 8..15 at K = 16+half*8+0..7. colptr = Bt[col] row (N x K).
__device__ __forceinline__ v16h frag_B(const _Float16* colptr, int k0, int half) {
  HFrag f;
  f.q[0] = *(const u32x4*)(colptr + k0 + half * 8);
  f.q[1] = *(const u32x4*)(colptr + k0 + 16 + half * 8);
  return f.v;
}

__device__ __forceinline__ float gelu_exact(float x) {
  return 0.5f * x * (1.0f + erff(x * 0.70710678118654752f));
}

// ---------------------------------------------------------------------------
// TDM: DMA a 2D f16 tile (tile_d1 rows x tile_d0 elems, row stride stride0
// elems in global) into LDS at lds_byte_addr. LDS padding: +4 DWORDs (8 f16)
// after every 16 DWORDs (32 f16 = one row) -> LDS row stride of 40 f16.
// D# layout per CDNA5 ISA. Wave-level op
// (EXEC ignored), tracked by TENSORcnt. Requires tile_d0 == 32.
// This toolchain exposes the 6-arg builtin:
//   (uint32x4 g0, int32x8 g1, int32x4 g2, int32x4 g3, int32x8, i32 cpol)
// ---------------------------------------------------------------------------
__device__ __forceinline__ void tdm_load_tile_f16(
    unsigned lds_byte_addr, const _Float16* gptr, unsigned tensor_d0,
    unsigned tensor_d1, unsigned tile_d0, unsigned tile_d1, unsigned stride0) {
#if USE_TDM
  unsigned long long ga = (unsigned long long)(uintptr_t)gptr;
  u32x4 g0;
  g0[0] = 1u;                                   // count=1, user descriptor
  g0[1] = lds_byte_addr;                        // lds_addr (bytes)
  g0[2] = (unsigned)(ga & 0xffffffffu);         // global_addr[31:0]
  g0[3] = (unsigned)((ga >> 32) & 0x01ffffffu)  // global_addr[56:32]
          | (2u << 30);                         // type = 2 ("image")
  i32x8 g1;
  g1[0] = (int)((1u << 16)      // data_size = 1 (2 bytes)
                | (1u << 20)    // pad_enable
                | (3u << 22)    // pad_interval: 2^(3+1)=16 DWORDs
                | (3u << 25));  // pad_amount: 4 DWORDs
  g1[1] = (int)((tensor_d0 & 0xffffu) << 16);                            // dim0 lo
  g1[2] = (int)(((tensor_d0 >> 16) & 0xffffu) | ((tensor_d1 & 0xffffu) << 16));
  g1[3] = (int)(((tensor_d1 >> 16) & 0xffffu) | (tile_d0 << 16));        // tile0
  g1[4] = (int)(tile_d1 & 0xffffu);                                      // tile1
  g1[5] = (int)stride0;  // tensor_dim0_stride lo32 (fits: K <= 4096)
  g1[6] = 0;
  g1[7] = 0;
  i32x4 z4 = {0, 0, 0, 0};
  i32x8 z8 = {0, 0, 0, 0, 0, 0, 0, 0};
  __builtin_amdgcn_tensor_load_to_lds(g0, g1, z4, z4, z8, 0);
#else
  (void)lds_byte_addr; (void)gptr; (void)tensor_d0; (void)tensor_d1;
  (void)tile_d0; (void)tile_d1; (void)stride0;
#endif
}

// ---------------------------------------------------------------------------
// fp32 -> f16 conversion (vectorized by 4)
// ---------------------------------------------------------------------------
__global__ void cvt_f32_f16(const float* __restrict__ src,
                            _Float16* __restrict__ dst, long n) {
  long i = ((long)blockIdx.x * blockDim.x + threadIdx.x) * 4;
  if (i >= n) return;
  float4 v = *(const float4*)(src + i);
  dst[i + 0] = (_Float16)v.x;
  dst[i + 1] = (_Float16)v.y;
  dst[i + 2] = (_Float16)v.z;
  dst[i + 3] = (_Float16)v.w;
}

// ---------------------------------------------------------------------------
// Tiled WMMA GEMM: C[M,N] = A[M,K] @ W[K,N] + bias, optional GELU.
// 256 threads = 8 waves (4x2). Workgroup tile 128x128, wave tile 32x64,
// K-step 32. Double-buffered LDS; A tile staged by TDM (wave 0), B tile
// staged transposed via VGPR path. ACT: 0=none, 1=gelu.
// OMODE: 0 = row-major [M,N]; 1 = head-split [B,H,S,HD] (N==1024,H=16,HD=64).
// ---------------------------------------------------------------------------
#define GTM 128
#define GTN 128
#define GTK 32
#define LDK 40  // padded K stride in LDS (80B rows, 16B aligned)

template <int ACT, int OMODE>
__global__ __launch_bounds__(256, 2) void gemm_f16(
    const _Float16* __restrict__ A, const _Float16* __restrict__ W,
    const float* __restrict__ bias, float* __restrict__ outF,
    _Float16* __restrict__ outH, int M, int N, int K, int S) {
  __shared__ _Float16 sA[2][GTM][LDK];
  __shared__ _Float16 sB[2][GTN][LDK];  // Bt: [n][k]

  const int tid  = threadIdx.x;
  const int lane = tid & 31;
  const int wave = tid >> 5;
  const int wm   = wave >> 1;  // 0..3
  const int wn   = wave & 1;   // 0..1
  const int r    = lane & 15;
  const int half = lane >> 4;

  const int m0 = blockIdx.y * GTM;
  const int n0 = blockIdx.x * GTN;

  v8f acc[2][4];
#pragma unroll
  for (int i = 0; i < 2; i++)
#pragma unroll
    for (int j = 0; j < 4; j++) {
      v8f z = {0.f, 0.f, 0.f, 0.f, 0.f, 0.f, 0.f, 0.f};
      acc[i][j] = z;
    }

  // Stage Bt tile (transposed into LDS via VGPR path; TDM cannot transpose)
  auto stageB = [&](int k0s, int buf) {
#pragma unroll
    for (int i = 0; i < 2; i++) {
      int c  = tid + i * 256;  // 0..511
      int kk = c >> 4;         // 0..31
      int nn = (c & 15) << 3;  // 0..120
      _Float16 tmp[8];
      *(u32x4*)tmp = *(const u32x4*)&W[(size_t)(k0s + kk) * N + n0 + nn];
#pragma unroll
      for (int j = 0; j < 8; j++) sB[buf][nn + j][kk] = tmp[j];
    }
    if (k0s + GTK < K)
      __builtin_prefetch(&W[(size_t)(k0s + GTK + (tid >> 4)) * N + n0 +
                            ((tid & 15) << 3)], 0, 0);
  };

  // Stage A tile: TDM DMA (wave 0 only) or manual fallback
  auto stageA = [&](int k0s, int buf) {
#if USE_TDM
    if (wave == 0)
      tdm_load_tile_f16((unsigned)(uintptr_t)&sA[buf][0][0],
                        &A[(size_t)m0 * K + k0s], (unsigned)(K - k0s),
                        (unsigned)(M - m0), GTK, GTM, (unsigned)K);
#else
#pragma unroll
    for (int i = 0; i < 2; i++) {
      int c   = tid + i * 256;
      int row = c >> 2;
      int kc  = (c & 3) << 3;
      *(u32x4*)&sA[buf][row][kc] =
          *(const u32x4*)&A[(size_t)(m0 + row) * K + k0s + kc];
    }
#endif
  };

  // Prologue: stage first tiles, drain TDM, sync
  stageB(0, 0);
  stageA(0, 0);
#if USE_TDM
  if (wave == 0) __builtin_amdgcn_s_wait_tensorcnt(0);
#endif
  __syncthreads();

  for (int k0 = 0; k0 < K; k0 += GTK) {
    const int cur = (k0 >> 5) & 1;
    const int nxt = cur ^ 1;
    if (k0 + GTK < K) {
      stageB(k0 + GTK, nxt);   // overlaps with WMMA below
      stageA(k0 + GTK, nxt);   // TDM overlaps with WMMA below
    }

    v16h af[2], bf[4];
#pragma unroll
    for (int i = 0; i < 2; i++)
      af[i] = frag_A(&sA[cur][wm * 32 + i * 16 + r][0], 0, half);
#pragma unroll
    for (int j = 0; j < 4; j++)
      bf[j] = frag_B(&sB[cur][wn * 64 + j * 16 + r][0], 0, half);

#pragma unroll
    for (int i = 0; i < 2; i++)
#pragma unroll
      for (int j = 0; j < 4; j++)
        acc[i][j] = __builtin_amdgcn_wmma_f32_16x16x32_f16(
            false, af[i], false, bf[j], (short)0, acc[i][j], false, false);

#if USE_TDM
    if (wave == 0) __builtin_amdgcn_s_wait_tensorcnt(0);
#endif
    __syncthreads();
  }

  // Epilogue: bias + activation + store (f32 and/or f16)
#pragma unroll
  for (int i = 0; i < 2; i++) {
#pragma unroll
    for (int j = 0; j < 4; j++) {
      int col  = n0 + wn * 64 + j * 16 + r;
      float bv = bias ? bias[col] : 0.f;
#pragma unroll
      for (int e = 0; e < 8; e++) {
        int row = m0 + wm * 32 + i * 16 + e + half * 8;
        float v = acc[i][j][e] + bv;
        if (ACT == 1) v = gelu_exact(v);
        size_t oidx;
        if (OMODE == 0) {
          oidx = (size_t)row * N + col;
        } else {
          int b = row / S, s = row - (row / S) * S;
          int h = col >> 6, d = col & 63;
          oidx = (((size_t)(b * 16 + h)) * S + s) * 64 + d;
        }
        if (outF) outF[oidx] = v;
        if (outH) outH[oidx] = (_Float16)v;
      }
    }
  }
}

// ---------------------------------------------------------------------------
// Flash attention: one workgroup = 64 query rows of one (b,h); 4 waves,
// each wave owns 16 query rows. Key blocks of 64 staged to LDS.
// HD = 64, scale = 1/8. Output ctx is f16 [B, S, D] (heads re-merged).
// ---------------------------------------------------------------------------
__global__ __launch_bounds__(128, 2) void attn_flash(
    const _Float16* __restrict__ Q, const _Float16* __restrict__ Kt,
    const _Float16* __restrict__ V, _Float16* __restrict__ ctx, int S) {
  __shared__ _Float16 sK[64][72];     // [key][d]
  __shared__ _Float16 sVt[64][72];    // [d][key]
  __shared__ _Float16 sP[4][16][72];  // per-wave probs [row][key]

  const int tid  = threadIdx.x;
  const int lane = tid & 31;
  const int wave = tid >> 5;
  const int r    = lane & 15;
  const int half = lane >> 4;
  const int bh   = blockIdx.y;  // b*H + h
  const int q0   = blockIdx.x * 64 + wave * 16;

  const _Float16* Qb = Q + (size_t)bh * S * 64;
  const _Float16* Kb = Kt + (size_t)bh * S * 64;
  const _Float16* Vb = V + (size_t)bh * S * 64;

  // Q fragments (A operand): rows q0+r, K dim = HD = 64 -> 2 frags
  v16h qf[2];
  {
    const _Float16* qrow = Qb + (size_t)(q0 + r) * 64;
    qf[0] = frag_A(qrow, 0, half);
    qf[1] = frag_A(qrow, 32, half);
  }

  float m[8], l[8];
  v8f o[4];
#pragma unroll
  for (int e = 0; e < 8; e++) { m[e] = -1e30f; l[e] = 0.f; }
#pragma unroll
  for (int j = 0; j < 4; j++) {
    v8f z = {0.f, 0.f, 0.f, 0.f, 0.f, 0.f, 0.f, 0.f};
    o[j] = z;
  }
  const float scale = 0.125f;  // 1/sqrt(64)

  for (int kb = 0; kb < S; kb += 64) {
    // Stage K block (as-is) and V block (transposed) into LDS
#pragma unroll
    for (int i = 0; i < 4; i++) {
      int c  = tid + i * 128;  // 0..511
      int kr = c >> 3;         // 0..63
      int d8 = (c & 7) << 3;   // 0..56
      *(u32x4*)&sK[kr][d8] = *(const u32x4*)&Kb[(size_t)(kb + kr) * 64 + d8];
      _Float16 tmp[8];
      *(u32x4*)tmp = *(const u32x4*)&Vb[(size_t)(kb + kr) * 64 + d8];
#pragma unroll
      for (int j = 0; j < 8; j++) sVt[d8 + j][kr] = tmp[j];
    }
    __syncthreads();

    // Scores: 16 (q) x 64 (keys) per wave, 4 tiles x 2 K-slices
    v8f sc[4];
#pragma unroll
    for (int j = 0; j < 4; j++) {
      v8f z = {0.f, 0.f, 0.f, 0.f, 0.f, 0.f, 0.f, 0.f};
      const _Float16* kc = &sK[j * 16 + r][0];
      z = __builtin_amdgcn_wmma_f32_16x16x32_f16(
          false, qf[0], false, frag_B(kc, 0, half), (short)0, z, false, false);
      z = __builtin_amdgcn_wmma_f32_16x16x32_f16(
          false, qf[1], false, frag_B(kc, 32, half), (short)0, z, false, false);
      sc[j] = z;
    }
#pragma unroll
    for (int j = 0; j < 4; j++)
#pragma unroll
      for (int e = 0; e < 8; e++) sc[j][e] *= scale;

    // Online softmax. C layout: row = e + 8*half, cols spread over the
    // 16 lanes of this half-wave -> butterfly over masks 8,4,2,1.
#pragma unroll
    for (int e = 0; e < 8; e++) {
      float mx = -1e30f;
#pragma unroll
      for (int j = 0; j < 4; j++) mx = fmaxf(mx, sc[j][e]);
      mx = fmaxf(mx, __shfl_xor(mx, 8, 32));
      mx = fmaxf(mx, __shfl_xor(mx, 4, 32));
      mx = fmaxf(mx, __shfl_xor(mx, 2, 32));
      mx = fmaxf(mx, __shfl_xor(mx, 1, 32));
      float mnew  = fmaxf(m[e], mx);
      float alpha = __expf(m[e] - mnew);
      float rs = 0.f;
#pragma unroll
      for (int j = 0; j < 4; j++) {
        float p = __expf(sc[j][e] - mnew);
        sc[j][e] = p;
        rs += p;
      }
      rs += __shfl_xor(rs, 8, 32);
      rs += __shfl_xor(rs, 4, 32);
      rs += __shfl_xor(rs, 2, 32);
      rs += __shfl_xor(rs, 1, 32);
      l[e] = l[e] * alpha + rs;
      m[e] = mnew;
#pragma unroll
      for (int j = 0; j < 4; j++) o[j][e] *= alpha;
      // Spill probs to LDS in [row][key] so they can be re-read as A frags
#pragma unroll
      for (int j = 0; j < 4; j++)
        sP[wave][e + 8 * half][j * 16 + r] = (_Float16)sc[j][e];
    }

    // O += P @ V  (P: 16x64, V: 64x64) — per-wave private sP region
#pragma unroll
    for (int ks = 0; ks < 2; ks++) {
      v16h pf = frag_A(&sP[wave][r][0], ks * 32, half);
#pragma unroll
      for (int j = 0; j < 4; j++) {
        v16h vf = frag_B(&sVt[j * 16 + r][0], ks * 32, half);
        o[j] = __builtin_amdgcn_wmma_f32_16x16x32_f16(
            false, pf, false, vf, (short)0, o[j], false, false);
      }
    }
    __syncthreads();
  }

  // Normalize and store ctx (heads merged): ctx[b][s][h*64 + d], f16
  const int b = bh >> 4, h = bh & 15;
#pragma unroll
  for (int e = 0; e < 8; e++) {
    int s = q0 + e + 8 * half;
    float inv = 1.f / l[e];
#pragma unroll
    for (int j = 0; j < 4; j++) {
      int d = h * 64 + j * 16 + r;
      ctx[((size_t)b * S + s) * 1024 + d] = (_Float16)(o[j][e] * inv);
    }
  }
}

// ---------------------------------------------------------------------------
// Fused residual add + LayerNorm (D=1024). One block per row, 256 threads.
// Writes f32 output and (optionally) an f16 copy for the next GEMM.
// ---------------------------------------------------------------------------
__global__ __launch_bounds__(256) void add_ln(
    const float* __restrict__ X, const float* __restrict__ R,
    const float* __restrict__ g, const float* __restrict__ be,
    float* __restrict__ outF, _Float16* __restrict__ outH) {
  const int row = blockIdx.x;
  const int tid = threadIdx.x;
  __shared__ float s1[256], s2[256];
  float x[4];
  float sum = 0.f, sq = 0.f;
#pragma unroll
  for (int i = 0; i < 4; i++) {
    int c   = tid + i * 256;
    float v = X[(size_t)row * 1024 + c] + R[(size_t)row * 1024 + c];
    x[i] = v;
    sum += v;
    sq  += v * v;
  }
  s1[tid] = sum;
  s2[tid] = sq;
  __syncthreads();
  for (int st = 128; st > 0; st >>= 1) {
    if (tid < st) {
      s1[tid] += s1[tid + st];
      s2[tid] += s2[tid + st];
    }
    __syncthreads();
  }
  float mu   = s1[0] * (1.f / 1024.f);
  float var  = s2[0] * (1.f / 1024.f) - mu * mu;
  float rstd = rsqrtf(var + 1e-5f);
#pragma unroll
  for (int i = 0; i < 4; i++) {
    int c   = tid + i * 256;
    float v = (x[i] - mu) * rstd * g[c] + be[c];
    if (outF) outF[(size_t)row * 1024 + c] = v;
    if (outH) outH[(size_t)row * 1024 + c] = (_Float16)v;
  }
}

// ---------------------------------------------------------------------------
// Launch
// ---------------------------------------------------------------------------
extern "C" void kernel_launch(void* const* d_in, const int* in_sizes, int n_in,
                              void* d_out, int out_size, void* d_ws,
                              size_t ws_size, hipStream_t stream) {
  (void)in_sizes; (void)n_in; (void)out_size; (void)ws_size;
  const int B = 4, S = 2048, D = 1024, DFF = 4096;
  const int M = B * S;  // 8192

  const float* x1  = (const float*)d_in[0];
  const float* x2  = (const float*)d_in[1];
  const float* Wq  = (const float*)d_in[2];
  const float* bq  = (const float*)d_in[3];
  const float* Wk  = (const float*)d_in[4];
  const float* bk  = (const float*)d_in[5];
  const float* Wv  = (const float*)d_in[6];
  const float* bv  = (const float*)d_in[7];
  const float* Wo  = (const float*)d_in[8];
  const float* bo  = (const float*)d_in[9];
  const float* g1  = (const float*)d_in[10];
  const float* b1  = (const float*)d_in[11];
  const float* g2  = (const float*)d_in[12];
  const float* b2  = (const float*)d_in[13];
  const float* Wf1 = (const float*)d_in[14];
  const float* bf1 = (const float*)d_in[15];
  const float* Wf2 = (const float*)d_in[16];
  const float* bf2 = (const float*)d_in[17];

  char* ws = (char*)d_ws;
  size_t off = 0;
  auto take = [&](size_t bytes) -> char* {
    char* p = ws + off;
    off = (off + bytes + 255) & ~(size_t)255;
    return p;
  };

  _Float16* x1h  = (_Float16*)take((size_t)M * D * 2);
  _Float16* x2h  = (_Float16*)take((size_t)M * D * 2);
  _Float16* Wqh  = (_Float16*)take((size_t)D * D * 2);
  _Float16* Wkh  = (_Float16*)take((size_t)D * D * 2);
  _Float16* Wvh  = (_Float16*)take((size_t)D * D * 2);
  _Float16* Woh  = (_Float16*)take((size_t)D * D * 2);
  _Float16* Wf1h = (_Float16*)take((size_t)D * DFF * 2);
  _Float16* Wf2h = (_Float16*)take((size_t)DFF * D * 2);
  _Float16* Qh   = (_Float16*)take((size_t)M * D * 2);  // [B,H,S,HD]
  _Float16* Kh   = (_Float16*)take((size_t)M * D * 2);
  _Float16* Vh   = (_Float16*)take((size_t)M * D * 2);
  _Float16* ctxh = (_Float16*)take((size_t)M * D * 2);  // [B,S,D]
  float*    atto = (float*)take((size_t)M * D * 4);
  float*    hbuf = (float*)take((size_t)M * D * 4);
  _Float16* hh   = (_Float16*)take((size_t)M * D * 2);
  _Float16* act  = (_Float16*)take((size_t)M * DFF * 2);
  float*    ffn  = (float*)take((size_t)M * D * 4);

  auto cvt = [&](const float* s, _Float16* d, long n) {
    cvt_f32_f16<<<(int)((n / 4 + 255) / 256), 256, 0, stream>>>(s, d, n);
  };
  cvt(x1, x1h, (long)M * D);
  cvt(x2, x2h, (long)M * D);
  cvt(Wq, Wqh, (long)D * D);
  cvt(Wk, Wkh, (long)D * D);
  cvt(Wv, Wvh, (long)D * D);
  cvt(Wo, Woh, (long)D * D);
  cvt(Wf1, Wf1h, (long)D * DFF);
  cvt(Wf2, Wf2h, (long)DFF * D);

  dim3 gq(D / GTN, M / GTM);  // (8, 64)
  gemm_f16<0, 1><<<gq, 256, 0, stream>>>(x1h, Wqh, bq, nullptr, Qh, M, D, D, S);
  gemm_f16<0, 1><<<gq, 256, 0, stream>>>(x2h, Wkh, bk, nullptr, Kh, M, D, D, S);
  gemm_f16<0, 1><<<gq, 256, 0, stream>>>(x2h, Wvh, bv, nullptr, Vh, M, D, D, S);

  dim3 ga(S / 64, B * 16);  // (32, 64)
  attn_flash<<<ga, 128, 0, stream>>>(Qh, Kh, Vh, ctxh, S);

  gemm_f16<0, 0><<<gq, 256, 0, stream>>>(ctxh, Woh, bo, atto, nullptr, M, D, D, S);
  add_ln<<<M, 256, 0, stream>>>(x1, atto, g1, b1, hbuf, hh);

  dim3 gf1(DFF / GTN, M / GTM);  // (32, 64)
  gemm_f16<1, 0><<<gf1, 256, 0, stream>>>(hh, Wf1h, bf1, nullptr, act, M, DFF, D, S);
  dim3 gf2(D / GTN, M / GTM);  // (8, 64)
  gemm_f16<0, 0><<<gf2, 256, 0, stream>>>(act, Wf2h, bf2, ffn, nullptr, M, D, DFF, S);

  add_ln<<<M, 256, 0, stream>>>(hbuf, ffn, g2, b2, (float*)d_out, nullptr);
}
